// MultiHeadAttention_56143812493518
// MI455X (gfx1250) — matrix operand order repaired and checked by reading, not compiled, not run on hardware
//
#include <hip/hip_runtime.h>
#include <hip/hip_bf16.h>

typedef __attribute__((ext_vector_type(16))) _Float16 v16h;
typedef __attribute__((ext_vector_type(8)))  _Float16 v8h;
typedef __attribute__((ext_vector_type(8)))  float    v8f;
typedef __attribute__((ext_vector_type(4)))  float    v4f;

#define WMMA_F16(a, b, c) \
  __builtin_amdgcn_wmma_f32_16x16x32_f16(false, (a), false, (b), (short)0, (c), false, false)

// A-fragment (16x32 f16): lane row = lane%16; elems 0..7 = K[h*8 .. h*8+7],
// elems 8..15 = K[16+h*8 .. 16+h*8+7]  (h = lane/16). Two b128 LDS loads.
static __device__ inline v16h ld_afrag(const _Float16* rowp, int halfid) {
  v8h lo = *(const v8h*)(rowp + halfid * 8);
  v8h hi = *(const v8h*)(rowp + 16 + halfid * 8);
  v16h r;
#pragma unroll
  for (int i = 0; i < 8; ++i) { r[i] = lo[i]; r[i + 8] = hi[i]; }
  return r;
}

// B-fragment (32x16 f16): lane col = lane%16; elems 0..7 -> K = h*8 + e, elems 8..15 -> K = 16 + h*8 + (e-8). Two b128 loads.
static __device__ inline v16h ld_bfrag(const _Float16* rowp, int halfid) {
  v8h lo = *(const v8h*)(rowp + halfid * 8);
  v8h hi = *(const v8h*)(rowp + 16 + halfid * 8);
  v16h r;
#pragma unroll
  for (int i = 0; i < 8; ++i) { r[i] = lo[i]; r[i + 8] = hi[i]; }
  return r;
}

// ---------------------------------------------------------------------------
// GEMM: O[M,N] = A[M,K] @ W[N,K]^T + bias[N]
// Block tile 128x128, k-step 32, 256 threads = 8 waves (4x2), wave tile 32x64.
// Register double-buffered: tile t+1 staged in VGPRs while WMMAs eat tile t.
// ---------------------------------------------------------------------------
constexpr int BM = 128, BN = 128, BK = 32, LDT = BK + 8;   // LDS row stride (pad)
constexpr int GS = 2048, GH = 16, GD = 64, GE = 1024;

template <bool AF16, bool HEADOUT>
__global__ __launch_bounds__(256) void gemm_kernel(
    const void* __restrict__ Av, const float* __restrict__ W,
    const float* __restrict__ bias, void* __restrict__ Ov,
    const int M, const int N, const int K) {
  __shared__ _Float16 As[BM * LDT];
  __shared__ _Float16 Bs[BN * LDT];
  const int tid = threadIdx.x;
  const int wv = tid >> 5, lane = tid & 31, halfid = lane >> 4, ln = lane & 15;
  const int wm = wv & 3, wn = wv >> 2;
  const int m0 = blockIdx.x * BM, n0 = blockIdx.y * BN;
  const int lrow = tid >> 1, lk = (tid & 1) * 16;

  struct Tile { v8h a0, a1, b0, b1; };
  auto stage = [&](int k0) -> Tile {
    Tile t;
    if constexpr (AF16) {
      const _Float16* A = (const _Float16*)Av;
      const v8h* src = (const v8h*)(A + (size_t)(m0 + lrow) * K + k0 + lk);
      t.a0 = src[0]; t.a1 = src[1];
    } else {
      const float* A = (const float*)Av;
      const v4f* src = (const v4f*)(A + (size_t)(m0 + lrow) * K + k0 + lk);
      v4f x0 = src[0], x1 = src[1], x2 = src[2], x3 = src[3];
#pragma unroll
      for (int i = 0; i < 4; ++i) {
        t.a0[i] = (_Float16)x0[i]; t.a0[4 + i] = (_Float16)x1[i];
        t.a1[i] = (_Float16)x2[i]; t.a1[4 + i] = (_Float16)x3[i];
      }
    }
    const v4f* ws = (const v4f*)(W + (size_t)(n0 + lrow) * K + k0 + lk);
    v4f y0 = ws[0], y1 = ws[1], y2 = ws[2], y3 = ws[3];
#pragma unroll
    for (int i = 0; i < 4; ++i) {
      t.b0[i] = (_Float16)y0[i]; t.b0[4 + i] = (_Float16)y1[i];
      t.b1[i] = (_Float16)y2[i]; t.b1[4 + i] = (_Float16)y3[i];
    }
    return t;
  };

  v8f acc[2][4];
#pragma unroll
  for (int i = 0; i < 2; ++i)
#pragma unroll
    for (int j = 0; j < 4; ++j) acc[i][j] = (v8f){0, 0, 0, 0, 0, 0, 0, 0};

  Tile cur = stage(0);
  const int nIter = K / BK;
  for (int it = 0; it < nIter; ++it) {
    *(v8h*)(&As[lrow * LDT + lk])     = cur.a0;
    *(v8h*)(&As[lrow * LDT + lk + 8]) = cur.a1;
    *(v8h*)(&Bs[lrow * LDT + lk])     = cur.b0;
    *(v8h*)(&Bs[lrow * LDT + lk + 8]) = cur.b1;
    __syncthreads();

    Tile nxt = cur;
    if (it + 1 < nIter) {
      nxt = stage((it + 1) * BK);                         // overlaps WMMAs below
      if (it + 2 < nIter) {
        const int kpf = (it + 2) * BK;
        if constexpr (AF16)
          __builtin_prefetch((const _Float16*)Av + (size_t)(m0 + lrow) * K + kpf + lk, 0, 3);
        else
          __builtin_prefetch((const float*)Av + (size_t)(m0 + lrow) * K + kpf + lk, 0, 3);
        __builtin_prefetch(W + (size_t)(n0 + lrow) * K + kpf + lk, 0, 3);
      }
    }

    v16h af[2], bf[4];
#pragma unroll
    for (int mt = 0; mt < 2; ++mt)
      af[mt] = ld_afrag(&As[(wm * 32 + mt * 16 + ln) * LDT], halfid);
#pragma unroll
    for (int nt = 0; nt < 4; ++nt)
      bf[nt] = ld_bfrag(&Bs[(wn * 64 + nt * 16 + ln) * LDT], halfid);
#pragma unroll
    for (int mt = 0; mt < 2; ++mt)
#pragma unroll
      for (int nt = 0; nt < 4; ++nt)
        acc[mt][nt] = WMMA_F16(af[mt], bf[nt], acc[mt][nt]);
    __syncthreads();
    cur = nxt;
  }

  // ---- epilogue: C frag row M = r + 8*halfid, col N = lane%16 ----
#pragma unroll
  for (int nt = 0; nt < 4; ++nt) {
    const int n = n0 + wn * 64 + nt * 16 + ln;
    const float bv = bias[n];
#pragma unroll
    for (int mt = 0; mt < 2; ++mt) {
#pragma unroll
      for (int r = 0; r < 8; ++r) {
        const int m = m0 + wm * 32 + mt * 16 + halfid * 8 + r;
        const float val = acc[mt][nt][r] + bv;
        if constexpr (HEADOUT) {
          _Float16* O = (_Float16*)Ov;
          const int bb = m >> 11, s = m & (GS - 1);
          const int hh = n >> 6, d = n & (GD - 1);
          O[(((size_t)bb * GH + hh) * GS + s) * GD + d] = (_Float16)val;
        } else {
          float* O = (float*)Ov;
          O[(size_t)m * N + n] = val;
        }
      }
    }
  }
}

// ---------------------------------------------------------------------------
// Fused flash attention. Q,K,V f16 in [B,H,S,D]; ctx f16 out [B,S,H*D].
// Double-buffered K/V LDS tiles; K tile fetched with GLOBAL_LOAD_ASYNC_TO_LDS
// (ASYNCcnt), V transposed through VGPRs; single barrier per 64-key tile.
// ---------------------------------------------------------------------------
__global__ __launch_bounds__(256) void attn_kernel(
    const _Float16* __restrict__ Q, const _Float16* __restrict__ Kp,
    const _Float16* __restrict__ V, _Float16* __restrict__ ctx) {
  constexpr int S = GS, D = GD, H = GH, LD = 72;
  constexpr int TILE = 64 * LD;                       // 4608 halfs
  __shared__ _Float16 lds[4 * TILE + 8 * 16 * LD];    // K0|K1|V0|V1|P  (55 KB)
  // Ping-pong buffers computed arithmetically (no pointer arrays: avoids
  // addrspacecast constant-aggregate initializers that ld.lld rejects).
  _Float16* Pb = lds + 4 * TILE;

  const int tid = threadIdx.x;
  const int wv = tid >> 5, lane = tid & 31, halfid = lane >> 4, ln = lane & 15;
  const int b = blockIdx.z, h = blockIdx.y;
  const int q0 = blockIdx.x * 128;
  const _Float16* Qh = Q + ((size_t)b * H + h) * S * D;
  const _Float16* Kh = Kp + ((size_t)b * H + h) * S * D;
  const _Float16* Vh = V + ((size_t)b * H + h) * S * D;

  const int krow = tid >> 2, kcs = (tid & 3) * 16;    // K/V staging coords

  // ---- stage Q (128x64) through the K0|K1 area, grab per-wave A-frags ----
  {
    const int row = tid >> 1, cs = (tid & 1) * 32;
    const v8h* src = (const v8h*)(Qh + (size_t)(q0 + row) * D + cs);
#pragma unroll
    for (int j = 0; j < 4; ++j) *(v8h*)(&lds[row * LD + cs + j * 8]) = src[j];
  }
  __syncthreads();
  v16h qa[2];
#pragma unroll
  for (int f = 0; f < 2; ++f)
    qa[f] = ld_afrag(&lds[(wv * 16 + ln) * LD + f * 32], halfid);
  __syncthreads();  // Q-frag reads done before tile 0 overwrites the area

  auto asyncK = [&](int kb0, _Float16* dst) {
    const unsigned lk2 = (unsigned)(size_t)(dst + krow * LD + kcs);
    const unsigned long long ga =
        (unsigned long long)(size_t)(Kh + (size_t)(kb0 + krow) * D + kcs);
    asm volatile("global_load_async_to_lds_b128 %0, %1, off\n\t"
                 "global_load_async_to_lds_b128 %0, %1, off offset:16"
                 :: "v"(lk2), "v"(ga) : "memory");
  };
  auto loadV = [&](int kb0, v8h& va, v8h& vb2) {
    const v8h* vs = (const v8h*)(Vh + (size_t)(kb0 + krow) * D + kcs);
    va = vs[0]; vb2 = vs[1];
  };
  auto storeVt = [&](_Float16* dst, v8h va, v8h vb2) {
#pragma unroll
    for (int i = 0; i < 8; ++i) {
      dst[(kcs + i) * LD + krow]     = va[i];
      dst[(kcs + 8 + i) * LD + krow] = vb2[i];
    }
  };

  // ---- preload tile 0 ----
  asyncK(0, lds);
  { v8h va, vb2; loadV(0, va, vb2); storeVt(lds + 2 * TILE, va, vb2); }
  asm volatile("s_wait_asynccnt 0x0" ::: "memory");
  __syncthreads();

  v8f o[4];
#pragma unroll
  for (int i = 0; i < 4; ++i) o[i] = (v8f){0, 0, 0, 0, 0, 0, 0, 0};
  float mrow[8], lrow[8];
#pragma unroll
  for (int r = 0; r < 8; ++r) { mrow[r] = -1e30f; lrow[r] = 0.f; }

  _Float16* Pw = Pb + wv * 16 * LD;
  const float scale = 0.125f;  // 1/sqrt(64)
  constexpr int NT = S / 64;

  for (int it = 0; it < NT; ++it) {
    const int p = it & 1;
    const _Float16* Ks = lds + p * TILE;
    const _Float16* Vt = lds + (2 + p) * TILE;

    // kick off next tile: async K -> other LDS buffer, V -> VGPRs
    v8h nva, nvb;
    const bool more = (it + 1 < NT);
    if (more) {
      asyncK((it + 1) * 64, lds + (p ^ 1) * TILE);
      loadV((it + 1) * 64, nva, nvb);
      if (it + 2 < NT) {
        __builtin_prefetch(Kh + (size_t)((it + 2) * 64 + krow) * D + kcs, 0, 3);
        __builtin_prefetch(Vh + (size_t)((it + 2) * 64 + krow) * D + kcs, 0, 3);
      }
    }

    // scores: S_blk[16x64] = Q(16xD) @ K_blk^T
    v8f sf[4];
#pragma unroll
    for (int nt = 0; nt < 4; ++nt) {
      v16h b0 = ld_bfrag(&Ks[(nt * 16 + ln) * LD + 0], halfid);
      v16h b1 = ld_bfrag(&Ks[(nt * 16 + ln) * LD + 32], halfid);
      v8f s = (v8f){0, 0, 0, 0, 0, 0, 0, 0};
      s = WMMA_F16(qa[0], b0, s);
      s = WMMA_F16(qa[1], b1, s);
#pragma unroll
      for (int r = 0; r < 8; ++r) s[r] *= scale;
      sf[nt] = s;
    }

    // online softmax (wave row = halfid*8 + r, striped over 16 lanes)
#pragma unroll
    for (int r = 0; r < 8; ++r) {
      float v = sf[0][r];
#pragma unroll
      for (int nt = 1; nt < 4; ++nt) v = fmaxf(v, sf[nt][r]);
#pragma unroll
      for (int off = 1; off < 16; off <<= 1) v = fmaxf(v, __shfl_xor(v, off, 32));
      const float mo = mrow[r];
      const float mn = fmaxf(mo, v);
      const float alpha = __expf(mo - mn);
      mrow[r] = mn;
#pragma unroll
      for (int dt = 0; dt < 4; ++dt) o[dt][r] *= alpha;
      float psum = 0.f;
#pragma unroll
      for (int nt = 0; nt < 4; ++nt) {
        const float pv = __expf(sf[nt][r] - mn);
        psum += pv;
        Pw[(halfid * 8 + r) * LD + nt * 16 + ln] = (_Float16)pv;  // C->A relayout
      }
#pragma unroll
      for (int off = 1; off < 16; off <<= 1) psum += __shfl_xor(psum, off, 32);
      lrow[r] = lrow[r] * alpha + psum;
    }
    // per-wave P buffer: same-wave LDS ordering, no block barrier needed
    asm volatile("s_wait_dscnt 0x0" ::: "memory");

    // O += P(16x64) @ V_blk(64x64)
    v16h pa0 = ld_afrag(&Pw[ln * LD + 0], halfid);
    v16h pa1 = ld_afrag(&Pw[ln * LD + 32], halfid);
#pragma unroll
    for (int dt = 0; dt < 4; ++dt) {
      v16h vb0 = ld_bfrag(&Vt[(dt * 16 + ln) * LD + 0], halfid);
      v16h vb1 = ld_bfrag(&Vt[(dt * 16 + ln) * LD + 32], halfid);
      o[dt] = WMMA_F16(pa0, vb0, o[dt]);
      o[dt] = WMMA_F16(pa1, vb1, o[dt]);
    }

    if (more) storeVt(lds + (2 + (p ^ 1)) * TILE, nva, nvb);  // other buffer
    asm volatile("s_wait_asynccnt 0x0" ::: "memory");         // next K tile landed
    __syncthreads();                                          // one barrier per tile
  }

  // normalize and write ctx [B, S, H*D]
#pragma unroll
  for (int dt = 0; dt < 4; ++dt) {
#pragma unroll
    for (int r = 0; r < 8; ++r) {
      const int qr = q0 + wv * 16 + halfid * 8 + r;
      const float val = o[dt][r] / lrow[r];
      ctx[((size_t)b * S + qr) * GE + h * D + dt * 16 + ln] = (_Float16)val;
    }
  }
}

// ---------------------------------------------------------------------------
extern "C" void kernel_launch(void* const* d_in, const int* in_sizes, int n_in,
                              void* d_out, int out_size, void* d_ws, size_t ws_size,
                              hipStream_t stream) {
  const float* query = (const float*)d_in[0];
  const float* key   = (const float*)d_in[1];
  const float* value = (const float*)d_in[2];
  const float* Wq = (const float*)d_in[3];
  const float* bq = (const float*)d_in[4];
  const float* Wk = (const float*)d_in[5];
  const float* bk = (const float*)d_in[6];
  const float* Wv = (const float*)d_in[7];
  const float* bv = (const float*)d_in[8];
  const float* Wo = (const float*)d_in[9];
  const float* bo = (const float*)d_in[10];

  const int B = 2, S = GS, E = GE, M = B * S;
  _Float16* ws = (_Float16*)d_ws;
  const size_t qkvElems = (size_t)B * GH * S * GD;  // 8 MB each as f16
  _Float16* qh = ws;
  _Float16* kh = ws + qkvElems;
  _Float16* vh = ws + 2 * qkvElems;
  _Float16* ch = ws + 3 * qkvElems;

  dim3 gg(M / BM, E / BN);
  gemm_kernel<false, true><<<gg, 256, 0, stream>>>(query, Wq, bq, qh, M, E, E);
  gemm_kernel<false, true><<<gg, 256, 0, stream>>>(key,   Wk, bk, kh, M, E, E);
  gemm_kernel<false, true><<<gg, 256, 0, stream>>>(value, Wv, bv, vh, M, E, E);
  attn_kernel<<<dim3(S / 128, GH, B), 256, 0, stream>>>(qh, kh, vh, ch);
  gemm_kernel<true, false><<<gg, 256, 0, stream>>>(ch, Wo, bo, d_out, M, E, E);
}
